// AttentionWithRoPE_6193342841628
// MI455X (gfx1250) — matrix operand order repaired and checked by reading, not compiled, not run on hardware
//
#include <hip/hip_runtime.h>
#include <hip/hip_bf16.h>

// ---------------------------------------------------------------------------
// AttentionWithRoPE for MI455X (gfx1250, wave32, WMMA bf16 path)
//   B=4, T=2048, DIM=1024, H=16, HD=64, SCALE = DIM^-0.5 = 1/32
// Pipeline:
//   1) x -> bf16 (vectorized)        xb   [M,K]
//   2) W* -> bf16 transposed (LDS)   W*t  [N,K]
//   3) 64x64-tile WMMA GEMM + RoPE -> qb (pre-scaled by ATT_SCALE), kb
//      64x64-tile WMMA GEMM        -> vt ([B,H,HD,T] bf16)
//   4) flash attention (WMMA + fused-DPP softmax reductions) -> ao bf16
//   5) 64x64-tile WMMA GEMM + bias -> d_out  fp32
// ---------------------------------------------------------------------------

typedef __attribute__((ext_vector_type(16))) __bf16 v16bf;
typedef __attribute__((ext_vector_type(8)))  float  v8f;

union FragAB {
    v16bf v;
    uint4 q[2];
};

__device__ __forceinline__ unsigned short f2bf(float f) {
    unsigned int u = __float_as_uint(f);
    unsigned int r = u + 0x7fffu + ((u >> 16) & 1u);   // round-to-nearest-even
    return (unsigned short)(r >> 16);
}

// ---- DPP16 cross-lane helpers (VALU only, no LDS) --------------------------
#define DPP_QUAD_XOR1 0xB1   // quad_perm [1,0,3,2]

template <int CTRL>
__device__ __forceinline__ float dpp_xfer(float v) {
    int r = __builtin_amdgcn_update_dpp(0, __float_as_int(v), CTRL, 0xF, 0xF, true);
    return __int_as_float(r);
}

// Plain hardware max (avoids clang's sNaN canonicalize around llvm.maxnum)
__device__ __forceinline__ float vmaxf(float a, float b) {
    float d;
    asm("v_max_num_f32_e32 %0, %1, %2" : "=v"(d) : "v"(a), "v"(b));
    return d;
}

// Fused butterfly steps: d = op(dpp(v), v) in ONE VALU instruction
// (DPP applies to src0; mnemonics match gfx1250 disassembler output).
#define DPP_RED_STEP(name, mnem, ctrl)                                   \
__device__ __forceinline__ float name(float v) {                         \
    float d;                                                             \
    asm(mnem " %0, %1, %1 " ctrl                                         \
        " row_mask:0xf bank_mask:0xf bound_ctrl:1"                       \
        : "=v"(d) : "v"(v));                                             \
    return d;                                                            \
}
DPP_RED_STEP(max_qx1, "v_max_num_f32_dpp", "quad_perm:[1,0,3,2]")
DPP_RED_STEP(max_qx2, "v_max_num_f32_dpp", "quad_perm:[2,3,0,1]")
DPP_RED_STEP(max_rm,  "v_max_num_f32_dpp", "row_mirror")
DPP_RED_STEP(max_rhm, "v_max_num_f32_dpp", "row_half_mirror")
DPP_RED_STEP(add_qx1, "v_add_f32_dpp",     "quad_perm:[1,0,3,2]")
DPP_RED_STEP(add_qx2, "v_add_f32_dpp",     "quad_perm:[2,3,0,1]")
DPP_RED_STEP(add_rm,  "v_add_f32_dpp",     "row_mirror")
DPP_RED_STEP(add_rhm, "v_add_f32_dpp",     "row_half_mirror")

// full reduction across each 16-lane half (all lanes receive the result)
__device__ __forceinline__ float row16_max(float v) {
    v = max_qx1(v);   // xor1 within quads
    v = max_qx2(v);   // xor2 -> quads reduced
    v = max_rm(v);    // q0<->q3, q1<->q2
    v = max_rhm(v);   // q0<->q1, q2<->q3
    return v;
}
__device__ __forceinline__ float row16_sum(float v) {
    v = add_qx1(v);
    v = add_qx2(v);
    v = add_rm(v);
    v = add_rhm(v);
    return v;
}

#define M_TOTAL   8192   // B*T
#define K_DIM     1024
#define N_DIM     1024
#define SEQ_T     2048
#define N_HEADS   16
#define HEAD_D    64
#define ATT_SCALE 0.03125f            // 1024^-0.5  (folded into Q projection)
#define LN_THETA  9.210340371976184f  // ln(10000)

// ---------------------------------------------------------------------------
// fp32 -> bf16, 4 elements / thread (b128 load, b64 store)
// ---------------------------------------------------------------------------
__global__ void cvt_bf16_kernel(const float* __restrict__ in,
                                unsigned int* __restrict__ out) {
    int i = blockIdx.x * blockDim.x + threadIdx.x;
    float4 f = ((const float4*)in)[i];
    unsigned int lo = (unsigned int)f2bf(f.x) | ((unsigned int)f2bf(f.y) << 16);
    unsigned int hi = (unsigned int)f2bf(f.z) | ((unsigned int)f2bf(f.w) << 16);
    ((uint2*)out)[i] = make_uint2(lo, hi);
}

// 1024x1024 weight: out[n*1024+k] = bf16(in[k*1024+n]); 32x32 LDS tile
__global__ void cvt_transpose_kernel(const float* __restrict__ in,
                                     unsigned short* __restrict__ out) {
    __shared__ float tile[32][33];
    const int bx = blockIdx.x & 31;   // n-tile
    const int by = blockIdx.x >> 5;   // k-tile
    const int tx = threadIdx.x & 31;
    const int ty = threadIdx.x >> 5;  // 0..7
    #pragma unroll
    for (int j = 0; j < 4; ++j) {
        int k = by * 32 + ty + j * 8;
        tile[ty + j * 8][tx] = in[k * 1024 + bx * 32 + tx];
    }
    __syncthreads();
    #pragma unroll
    for (int j = 0; j < 4; ++j) {
        int n = bx * 32 + ty + j * 8;
        out[(size_t)n * 1024 + by * 32 + tx] = f2bf(tile[tx][ty + j * 8]);
    }
}

// ---------------------------------------------------------------------------
// 64x64-tile bf16 WMMA GEMM: one wave owns a 64x64 fp32 tile of A[M,K]*Bt[N,K]^T
// 16 WMMAs per 16 b128 loads per k-step (4x reuse of both operands).
//   MODE 0: RoPE epilogue * ascale, bf16 -> [B,H,T,HD]   (Q: ascale=1/32, K: 1)
//   MODE 1: plain epilogue, bf16 -> [B,H,HD,T]           (V, transposed)
//   MODE 2: +bias, fp32 -> [M,N]                         (output projection)
// ---------------------------------------------------------------------------
template <int MODE>
__global__ void gemm64_kernel(const unsigned short* __restrict__ A,
                              const unsigned short* __restrict__ Bt,
                              void* __restrict__ dst,
                              const float* __restrict__ bias,
                              float ascale) {
    const int lane = threadIdx.x & 31;
    const int ln   = lane & 15;
    const int half = lane >> 4;
    const int wid  = blockIdx.x * (blockDim.x >> 5) + (threadIdx.x >> 5);

    const int tiles_n = N_DIM / 64;           // 16
    const int mt = wid / tiles_n;
    const int nt = wid % tiles_n;

    const unsigned short* arow[4];
    const unsigned short* brow[4];
    #pragma unroll
    for (int i = 0; i < 4; ++i) {
        arow[i] = A  + (size_t)(mt * 64 + i * 16 + ln) * K_DIM;
        brow[i] = Bt + (size_t)(nt * 64 + i * 16 + ln) * K_DIM;
    }

    v8f acc[16];
    #pragma unroll
    for (int i = 0; i < 16; ++i) acc[i] = (v8f){0.f,0.f,0.f,0.f,0.f,0.f,0.f,0.f};

    for (int k0 = 0; k0 < K_DIM; k0 += 32) {
        FragAB a[4], b[4];
        #pragma unroll
        for (int i = 0; i < 4; ++i) {
            __builtin_prefetch(arow[i] + k0 + 64, 0, 0);   // global_prefetch_b8
            __builtin_prefetch(brow[i] + k0 + 64, 0, 0);
            // A-layout: lanes 0-15 K 0..15, lanes 16-31 K 16..31
            a[i].q[0] = *(const uint4*)(arow[i] + k0 + half * 16);
            a[i].q[1] = *(const uint4*)(arow[i] + k0 + half * 16 + 8);
            // B-layout: lanes 0-15 K 0..15, lanes 16-31 K 16..31
            b[i].q[0] = *(const uint4*)(brow[i] + k0 + half * 16);
            b[i].q[1] = *(const uint4*)(brow[i] + k0 + half * 16 + 8);
        }
        #pragma unroll
        for (int mi = 0; mi < 4; ++mi)
            #pragma unroll
            for (int ni = 0; ni < 4; ++ni)
                acc[mi * 4 + ni] = __builtin_amdgcn_wmma_f32_16x16x32_bf16(
                    false, a[mi].v, false, b[ni].v, (short)0,
                    acc[mi * 4 + ni], false, false);
    }

    // C-layout: VGPR r -> row r + 8*half, col = ln (within each 16x16 subtile)
    #pragma unroll
    for (int mi = 0; mi < 4; ++mi) {
        #pragma unroll
        for (int ni = 0; ni < 4; ++ni) {
            #pragma unroll
            for (int r = 0; r < 8; ++r) {
                float val = acc[mi * 4 + ni][r];
                const int m = mt * 64 + mi * 16 + r + 8 * half;
                const int n = nt * 64 + ni * 16 + ln;

                if constexpr (MODE == 2) {
                    ((float*)dst)[(size_t)m * N_DIM + n] = val + bias[n];
                } else {
                    const int bidx = m >> 11;          // / 2048
                    const int t    = m & 2047;
                    const int h    = n >> 6;           // / 64
                    const int d    = n & 63;
                    if constexpr (MODE == 0) {
                        // RoPE pair = adjacent lane; DPP quad_perm xor1
                        float partner = dpp_xfer<DPP_QUAD_XOR1>(val);
                        float invf = __expf(-((float)(d & ~1) / 64.0f) * LN_THETA);
                        float ang  = (float)t * invf;
                        float c = __cosf(ang), s = __sinf(ang);
                        float rot = (d & 1) ? partner : -partner;
                        float o = (val * c + rot * s) * ascale;
                        size_t idx =
                            ((size_t)(bidx * N_HEADS + h) * SEQ_T + t) * HEAD_D + d;
                        ((unsigned short*)dst)[idx] = f2bf(o);
                    } else {  // MODE 1: V transposed [B,H,HD,T]
                        size_t idx =
                            ((size_t)(bidx * N_HEADS + h) * HEAD_D + d) * SEQ_T + t;
                        ((unsigned short*)dst)[idx] = f2bf(val);
                    }
                }
            }
        }
    }
}

// ---------------------------------------------------------------------------
// Flash attention: one wave per (b,h, 16-query block); 64 key tiles of 32.
// Q comes in pre-scaled by ATT_SCALE. Online softmax uses fused-DPP
// reductions; P transposed C-layout -> A-layout through a 1KB/wave LDS slab.
//   qb/kb: [B,H,T,HD] bf16   vt: [B,H,HD,T] bf16   ao: [M, DIM] bf16
// ---------------------------------------------------------------------------
__global__ void flash_attn_kernel(const unsigned short* __restrict__ qb,
                                  const unsigned short* __restrict__ kb,
                                  const unsigned short* __restrict__ vt,
                                  unsigned short* __restrict__ ao) {
    const int lane = threadIdx.x & 31;
    const int ln   = lane & 15;
    const int half = lane >> 4;
    const int w    = threadIdx.x >> 5;
    const int task = blockIdx.x * (blockDim.x >> 5) + w;   // 8192 tasks
    const int bh = task >> 7;     // b*16 + h
    const int qt = task & 127;    // query block

    __shared__ __align__(16) unsigned short Pl[4][16][32];

    // Q fragments (reused across all key tiles)
    const unsigned short* qrow =
        qb + ((size_t)bh * SEQ_T + qt * 16 + ln) * HEAD_D;
    FragAB aq0, aq1;
    aq0.q[0] = *(const uint4*)(qrow + half * 16);
    aq0.q[1] = *(const uint4*)(qrow + half * 16 + 8);
    aq1.q[0] = *(const uint4*)(qrow + 32 + half * 16);
    aq1.q[1] = *(const uint4*)(qrow + 32 + half * 16 + 8);

    float mrun[8], lrun[8];
    v8f o[4];
    #pragma unroll
    for (int r = 0; r < 8; ++r) { mrun[r] = -3.0e38f; lrun[r] = 0.f; }
    #pragma unroll
    for (int i = 0; i < 4; ++i) o[i] = (v8f){0.f,0.f,0.f,0.f,0.f,0.f,0.f,0.f};

    for (int kt = 0; kt < SEQ_T / 32; ++kt) {
        const int keybase = kt * 32;

        // ---- S = Q * K^T for two 16-key subtiles --------------------------
        const unsigned short* krow0 =
            kb + ((size_t)bh * SEQ_T + keybase + ln) * HEAD_D;
        const unsigned short* krow1 = krow0 + 16 * HEAD_D;

        v8f s0 = (v8f){0.f,0.f,0.f,0.f,0.f,0.f,0.f,0.f};
        v8f s1 = (v8f){0.f,0.f,0.f,0.f,0.f,0.f,0.f,0.f};
        FragAB b;

        b.q[0] = *(const uint4*)(krow0 + half * 16);
        b.q[1] = *(const uint4*)(krow0 + half * 16 + 8);
        s0 = __builtin_amdgcn_wmma_f32_16x16x32_bf16(false, aq0.v, false, b.v,
                                                     (short)0, s0, false, false);
        b.q[0] = *(const uint4*)(krow0 + 32 + half * 16);
        b.q[1] = *(const uint4*)(krow0 + 32 + half * 16 + 8);
        s0 = __builtin_amdgcn_wmma_f32_16x16x32_bf16(false, aq1.v, false, b.v,
                                                     (short)0, s0, false, false);
        b.q[0] = *(const uint4*)(krow1 + half * 16);
        b.q[1] = *(const uint4*)(krow1 + half * 16 + 8);
        s1 = __builtin_amdgcn_wmma_f32_16x16x32_bf16(false, aq0.v, false, b.v,
                                                     (short)0, s1, false, false);
        b.q[0] = *(const uint4*)(krow1 + 32 + half * 16);
        b.q[1] = *(const uint4*)(krow1 + 32 + half * 16 + 8);
        s1 = __builtin_amdgcn_wmma_f32_16x16x32_bf16(false, aq1.v, false, b.v,
                                                     (short)0, s1, false, false);

        // ---- online softmax, fused DPP16 reductions (no canonicalizes) ----
        #pragma unroll
        for (int r = 0; r < 8; ++r) {
            float x0 = s0[r];                 // Q pre-scaled by ATT_SCALE
            float x1 = s1[r];
            float tmax = row16_max(vmaxf(x0, x1));
            float nm   = vmaxf(mrun[r], tmax);
            float corr = __expf(mrun[r] - nm);
            float p0 = __expf(x0 - nm);
            float p1 = __expf(x1 - nm);
            float ps = row16_sum(p0 + p1);
            lrun[r] = lrun[r] * corr + ps;
            mrun[r] = nm;
            o[0][r] *= corr; o[1][r] *= corr; o[2][r] *= corr; o[3][r] *= corr;

            const int row = r + 8 * half;
            Pl[w][row][ln]      = f2bf(p0);
            Pl[w][row][16 + ln] = f2bf(p1);
        }
        __syncthreads();   // all 4 waves share identical trip count

        // ---- P (A-layout from LDS) * V ------------------------------------
        FragAB ap;
        ap.q[0] = *(const uint4*)(&Pl[w][ln][half * 16]);
        ap.q[1] = *(const uint4*)(&Pl[w][ln][half * 16 + 8]);

        #pragma unroll
        for (int nt = 0; nt < 4; ++nt) {
            const unsigned short* vrow =
                vt + ((size_t)bh * HEAD_D + nt * 16 + ln) * SEQ_T + keybase;
            FragAB bv;
            bv.q[0] = *(const uint4*)(vrow + half * 16);
            bv.q[1] = *(const uint4*)(vrow + half * 16 + 8);
            o[nt] = __builtin_amdgcn_wmma_f32_16x16x32_bf16(
                false, ap.v, false, bv.v, (short)0, o[nt], false, false);
        }
        __syncthreads();   // Pl reused next iteration
    }

    // ---- normalize (1 divide per row) and store bf16 ----------------------
    const int bb = bh >> 4, h = bh & 15;
    #pragma unroll
    for (int r = 0; r < 8; ++r) {
        const float inv = 1.0f / lrun[r];
        const int t = qt * 16 + r + 8 * half;
        #pragma unroll
        for (int nt = 0; nt < 4; ++nt) {
            const int col = h * HEAD_D + nt * 16 + ln;
            float val = o[nt][r] * inv;
            ao[(size_t)(bb * SEQ_T + t) * N_DIM + col] = f2bf(val);
        }
    }
}

// ---------------------------------------------------------------------------
// Host launch
// ---------------------------------------------------------------------------
extern "C" void kernel_launch(void* const* d_in, const int* in_sizes, int n_in,
                              void* d_out, int out_size, void* d_ws, size_t ws_size,
                              hipStream_t stream) {
    const float* x  = (const float*)d_in[0];
    // d_in[1]='h', d_in[2]='w' : unused by the reference math
    const float* Wq = (const float*)d_in[3];
    const float* Wk = (const float*)d_in[4];
    const float* Wv = (const float*)d_in[5];
    const float* Wp = (const float*)d_in[6];
    const float* bp = (const float*)d_in[7];

    char* ws = (char*)d_ws;
    const size_t SZ_X = (size_t)M_TOTAL * K_DIM * 2;   // 16 MB
    const size_t SZ_W = (size_t)N_DIM * K_DIM * 2;     //  2 MB
    unsigned short* xb  = (unsigned short*)(ws);
    unsigned short* Wqt = (unsigned short*)(ws + SZ_X);
    unsigned short* Wkt = (unsigned short*)(ws + SZ_X + SZ_W);
    unsigned short* Wvt = (unsigned short*)(ws + SZ_X + 2 * SZ_W);
    unsigned short* Wpt = (unsigned short*)(ws + SZ_X + 3 * SZ_W);
    unsigned short* qb  = (unsigned short*)(ws + SZ_X + 4 * SZ_W);
    unsigned short* kb  = (unsigned short*)(ws + 2 * SZ_X + 4 * SZ_W);
    unsigned short* vt  = (unsigned short*)(ws + 3 * SZ_X + 4 * SZ_W);
    unsigned short* ao  = (unsigned short*)(ws + 4 * SZ_X + 4 * SZ_W);

    // 1) conversions
    {
        int n4 = (M_TOTAL * K_DIM) / 4;   // 4 floats per thread
        cvt_bf16_kernel<<<n4 / 256, 256, 0, stream>>>(x, (unsigned int*)xb);
        cvt_transpose_kernel<<<1024, 256, 0, stream>>>(Wq, Wqt);
        cvt_transpose_kernel<<<1024, 256, 0, stream>>>(Wk, Wkt);
        cvt_transpose_kernel<<<1024, 256, 0, stream>>>(Wv, Wvt);
        cvt_transpose_kernel<<<1024, 256, 0, stream>>>(Wp, Wpt);
    }

    // 2) QKV projections: (8192/64)*(1024/64)=2048 waves, 4 waves/block
    const int gemm_blocks = (M_TOTAL / 64) * (N_DIM / 64) / 4;   // 512
    gemm64_kernel<0><<<gemm_blocks, 128, 0, stream>>>(xb, Wqt, qb, nullptr,
                                                      ATT_SCALE);   // Q pre-scaled
    gemm64_kernel<0><<<gemm_blocks, 128, 0, stream>>>(xb, Wkt, kb, nullptr, 1.0f);
    gemm64_kernel<1><<<gemm_blocks, 128, 0, stream>>>(xb, Wvt, vt, nullptr, 1.0f);

    // 3) attention: 64 (b,h) * 128 q-blocks = 8192 wave tasks -> 2048 blocks
    flash_attn_kernel<<<8192 / 4, 128, 0, stream>>>(qb, kb, vt, ao);

    // 4) output projection + bias -> fp32 d_out
    gemm64_kernel<2><<<gemm_blocks, 128, 0, stream>>>(ao, Wpt, d_out, bp, 1.0f);
}
